// NeRFModel_73856257622319
// MI455X (gfx1250) — compile-verified, matrix operand order checked by reading, not run on hardware
//
#include <hip/hip_runtime.h>
#include <math.h>

typedef __attribute__((ext_vector_type(16))) _Float16 v16h;
typedef __attribute__((ext_vector_type(8)))  float    v8f;

// pointee type exactly as the async-LDS builtin expects: gcc-vector int4
typedef int v4i_vs __attribute__((vector_size(16)));
typedef __attribute__((address_space(1))) v4i_vs* async_gptr;   // global src
typedef __attribute__((address_space(3))) v4i_vs* async_lptr;   // LDS dst

#define LAYER_W   256
#define LDSK      264          // padded LDS row stride in halves (528B = 16B-aligned rows)
#define MTILE     128          // points per workgroup
#define N_COARSE  64
#define N_FINE    128
#define N_RAYS    1024
#define PI_F      3.14159265358979f

// transposed-weight global layout (halves):
//   layer0 (K padded to 64):  [0, 16384)          Wt0[n*64 + k]
//   hidden l:                 16384 + l*65536     WtH[n*256 + k]
//   out:                      475136              WtO[n*256 + k]
#define WOFF_HID  16384
#define WOFF_OUT  (16384 + 7 * 65536)
#define NW_TOTAL  (16384 + 8 * 65536)   // 540672 halves

__device__ __forceinline__ void wait_async() {
  asm volatile("s_wait_asynccnt 0" ::: "memory");
}

__device__ __forceinline__ void async_copy16(const _Float16* gsrc, _Float16* ldst) {
  __builtin_amdgcn_global_load_async_to_lds_b128(
      (async_gptr)gsrc, (async_lptr)ldst, 0, 0);
}

// ---------------- fragment loader: two 16B LDS loads -> v16h -----------------
__device__ __forceinline__ v16h load_frag(const _Float16* p) {
  union { v16h v; uint4 q[2]; } u;
  u.q[0] = *(const uint4*)(p);        // halves 0..7   (K = base .. base+7)
  u.q[1] = *(const uint4*)(p + 16);   // halves 8..15  (K = base+16 .. base+23)
  return u.v;
}

// ------- async DMA staging: contiguous Wt[n][k] rows -> padded LDS rows ------
// copies 256 rows x kpad halves; kp_log8 = log2(kpad/8) chunks per row
__device__ __forceinline__ void stage_async(_Float16* wlds,
                                            const _Float16* __restrict__ wsrc,
                                            int kpad, int kp_log8, int tid) {
  const int chunks = (LAYER_W * kpad) >> 3;     // 16B chunks
  for (int idx = tid; idx < chunks; idx += 256) {
    int n = idx >> kp_log8;
    int k = (idx & ((1 << kp_log8) - 1)) << 3;
    async_copy16(wsrc + n * kpad + k, wlds + n * LDSK + k);
  }
}

// ---------------- one dense layer: h_out = act(h_in @ W + b) via WMMA --------
__device__ __forceinline__ void gemm_layer(const _Float16* hin, _Float16* hout,
                                           const _Float16* wlds,
                                           const float* __restrict__ bias,
                                           int K, bool relu,
                                           int r, int c, int msub) {
  const int KT = K >> 5;
  for (int half = 0; half < 2; ++half) {     // split 16 N-tiles into 2 passes of 8
    v8f acc[8] = {};
    for (int kt = 0; kt < KT; ++kt) {
      // A fragment: row (msub*16 + r), K-runs {kt*32+c*8 .. +7} and {+16..}
      v16h a = load_frag(hin + (msub * 16 + r) * LDSK + kt * 32 + c * 8);
#pragma unroll
      for (int nt = 0; nt < 8; ++nt) {
        int n = (half * 8 + nt) * 16 + r;    // B fragment: column n, same K-runs
        v16h b = load_frag(wlds + n * LDSK + kt * 32 + c * 8);
        acc[nt] = __builtin_amdgcn_wmma_f32_16x16x32_f16(
            false, a, false, b, (short)0, acc[nt], false, false);
      }
    }
#pragma unroll
    for (int nt = 0; nt < 8; ++nt) {
      int n = (half * 8 + nt) * 16 + r;
      float bv = bias[n];
#pragma unroll
      for (int j = 0; j < 8; ++j) {          // C layout: VGPR j -> M = j + 8*c
        float v = acc[nt][j] + bv;
        if (relu) v = fmaxf(v, 0.0f);
        hout[(msub * 16 + 8 * c + j) * LDSK + n] = (_Float16)v;
      }
    }
  }
}

// ---------------- fused MLP over a tile of 128 points ------------------------
__global__ void __launch_bounds__(256)
mlp_kernel(const _Float16* __restrict__ enc, const _Float16* __restrict__ wh,
           const float* __restrict__ b_in, const float* __restrict__ b_hid,
           const float* __restrict__ b_out,
           const float* __restrict__ W_sigma, const float* __restrict__ b_sigma,
           const float* __restrict__ W_color, const float* __restrict__ b_color,
           const float* __restrict__ trans,
           float* __restrict__ sigma_out, float* __restrict__ color_out,
           int npts) {
  extern __shared__ char smem[];
  _Float16* wlds  = (_Float16*)smem;                              // 256*LDSK halves
  _Float16* hbufA = (_Float16*)(smem + LAYER_W * LDSK * 2);       // 128*LDSK halves
  _Float16* hbufB = hbufA + MTILE * LDSK;

  const int tid  = threadIdx.x;
  const int lane = tid & 31;
  const int r    = lane & 15;
  const int c    = lane >> 4;
  const int msub = tid >> 5;
  const long long pbase = (long long)blockIdx.x * MTILE;

  // async DMA: encoded points (128 x 64 halves) -> h-buffer A, 16B chunks
  for (int idx = tid; idx < MTILE * 8; idx += 256) {
    int p = idx >> 3;
    int k = (idx & 7) * 8;
    async_copy16(enc + (pbase + p) * 64 + k, hbufA + p * LDSK + k);
  }

  _Float16* hin  = hbufA;
  _Float16* hout = hbufB;

  // layer 0: 60(->64 padded) -> 256, ReLU
  stage_async(wlds, wh, 64, 3, tid);
  wait_async();
  __syncthreads();
  gemm_layer(hin, hout, wlds, b_in, 64, true, r, c, msub);
  { _Float16* t = hin; hin = hout; hout = t; }

  // 7 hidden layers 256 -> 256, ReLU
  const _Float16* wp = wh + WOFF_HID;
  for (int l = 0; l < 7; ++l) {
    __syncthreads();
    stage_async(wlds, wp, 256, 5, tid);
    wait_async();
    __syncthreads();
    gemm_layer(hin, hout, wlds, b_hid + l * LAYER_W, 256, true, r, c, msub);
    { _Float16* t = hin; hin = hout; hout = t; }
    wp += LAYER_W * LAYER_W;
  }

  // output layer 256 -> 256, no activation
  __syncthreads();
  stage_async(wlds, wh + WOFF_OUT, 256, 5, tid);
  wait_async();
  __syncthreads();
  gemm_layer(hin, hout, wlds, b_out, 256, false, r, c, msub);
  { _Float16* t = hin; hin = hout; hout = t; }
  __syncthreads();

  // heads: sigma + view-dependent color (direction is constant per dispatch)
  if (tid < MTILE) {
    long long p = pbase + tid;
    if (p < npts) {
      const _Float16* h = hin + tid * LDSK;
      float s = b_sigma[0];
      for (int k = 0; k < 256; ++k) s += (float)h[k] * W_sigma[k];
      sigma_out[p] = 1.0f / (1.0f + expf(-s));

      float dw[3];
#pragma unroll
      for (int j = 0; j < 3; ++j) dw[j] = -trans[j * 4 + 2] + trans[j * 4 + 3];
      float de[24];
#pragma unroll
      for (int j = 0; j < 3; ++j)
#pragma unroll
        for (int l = 0; l < 4; ++l) {
          float f = exp2f((float)l * (4.0f / 3.0f)) * PI_F;
          float a = dw[j] * f;
          de[j * 8 + l * 2 + 0] = sinf(a);
          de[j * 8 + l * 2 + 1] = cosf(a);
        }
#pragma unroll
      for (int ch = 0; ch < 3; ++ch) {
        float acc = b_color[ch];
        for (int d = 0; d < 24; ++d) acc += de[d] * W_color[d * 3 + ch];
        for (int k = 0; k < 256; ++k) acc += (float)h[k] * W_color[(24 + k) * 3 + ch];
        color_out[p * 3 + ch] = 1.0f / (1.0f + expf(-acc));
      }
    }
  }
}

// ------------- weights fp32 -> fp16, pre-transposed per layer ---------------
__global__ void convert_w_kernel(const float* __restrict__ Win,
                                 const float* __restrict__ Whid,
                                 const float* __restrict__ Wout,
                                 _Float16* __restrict__ wh) {
  int i = blockIdx.x * blockDim.x + threadIdx.x;
  if (i >= NW_TOTAL) return;
  if (i < WOFF_HID) {                       // input layer, K padded 60 -> 64
    int n = i >> 6, k = i & 63;
    wh[i] = (k < 60) ? (_Float16)Win[k * LAYER_W + n] : (_Float16)0.0f;
  } else if (i < WOFF_OUT) {                // 7 hidden layers
    int j = i - WOFF_HID;
    int l = j >> 16, rr = j & 65535;
    int n = rr >> 8, k = rr & 255;
    wh[i] = (_Float16)Whid[l * 65536 + k * LAYER_W + n];
  } else {                                  // output layer
    int rr = i - WOFF_OUT;
    int n = rr >> 8, k = rr & 255;
    wh[i] = (_Float16)Wout[k * LAYER_W + n];
  }
}

// ---------------- positional encoding of world-space points -----------------
__global__ void prep_enc_kernel(const float* __restrict__ hor,
                                const float* __restrict__ ver,
                                const float* __restrict__ trans,
                                const float* __restrict__ tsrc,
                                const int* __restrict__ d_near,
                                const int* __restrict__ d_far,
                                _Float16* __restrict__ enc, int nspr, int npts) {
  int p = blockIdx.x * blockDim.x + threadIdx.x;
  if (p >= npts) return;
  int ray = p / nspr, s = p - ray * nspr;
  float nf = (float)d_near[0], ff = (float)d_far[0];
  float t = tsrc ? tsrc[p]
                 : (nf + (ff - nf) * (float)s / (float)(N_COARSE - 1));
  float cam[4] = { hor[ray], -ver[ray], t, 1.0f };
  _Float16* e = enc + (long long)p * 64;
#pragma unroll
  for (int j = 0; j < 3; ++j) {
    float x = cam[0] * trans[j * 4 + 0] + cam[1] * trans[j * 4 + 1] +
              cam[2] * trans[j * 4 + 2] + cam[3] * trans[j * 4 + 3];
#pragma unroll
    for (int l = 0; l < 10; ++l) {
      float f = exp2f((float)l * (10.0f / 9.0f)) * PI_F;
      float a = x * f;
      e[j * 20 + l * 2 + 0] = (_Float16)sinf(a);
      e[j * 20 + l * 2 + 1] = (_Float16)cosf(a);
    }
  }
  e[60] = e[61] = e[62] = e[63] = (_Float16)0.0f;
}

// ---------------- hierarchical inverse-CDF resampling -----------------------
__global__ void resample_kernel(const float* __restrict__ sigma_co,
                                const int* __restrict__ d_near,
                                const int* __restrict__ d_far,
                                float* __restrict__ t_fi) {
  int ray = blockIdx.x * blockDim.x + threadIdx.x;
  if (ray >= N_RAYS) return;
  float nf = (float)d_near[0], ff = (float)d_far[0];
  float sig[N_COARSE], cdf[N_COARSE];
  float run = 0.0f;
  for (int i = 0; i < N_COARSE; ++i) {
    sig[i] = sigma_co[ray * N_COARSE + i];
    run += sig[i];
    cdf[i] = run;
  }
  float low = cdf[0], high = cdf[N_COARSE - 1];
  float delta = (ff - nf) / (float)(N_COARSE - 1);
  for (int j = 0; j < N_FINE; ++j) {
    float ti = low + (high - low) * (float)(j + 1) / (float)(N_FINE + 1);
    int lo = 0, hi = N_COARSE;                       // searchsorted 'left'
    while (lo < hi) { int mid = (lo + hi) >> 1; if (cdf[mid] < ti) lo = mid + 1; else hi = mid; }
    int idx = lo - 1;
    idx = idx < 0 ? 0 : (idx > N_COARSE - 1 ? N_COARSE - 1 : idx);
    float slope = (idx < N_COARSE - 1) ? delta / sig[idx + 1] : 0.0f;
    float lt = nf + delta * (float)idx;
    t_fi[ray * N_FINE + j] = lt + (ti - cdf[idx]) * slope;
  }
}

// ---------------- merge + volume render -------------------------------------
__global__ void render_kernel(const float* __restrict__ sigma_co,
                              const float* __restrict__ color_co,
                              const float* __restrict__ sigma_fi,
                              const float* __restrict__ color_fi,
                              const float* __restrict__ t_fi,
                              const int* __restrict__ d_near,
                              const int* __restrict__ d_far,
                              float* __restrict__ out) {
  int ray = blockIdx.x * blockDim.x + threadIdx.x;
  if (ray >= N_RAYS) return;
  float nf = (float)d_near[0], ff = (float)d_far[0];

  // coarse pass: fixed delta = (far-near)/Nc, inclusive cumsum transmittance
  float dco = (ff - nf) / (float)N_COARSE;
  float cum = 0.0f, c0 = 0.0f, c1 = 0.0f, c2 = 0.0f;
  for (int i = 0; i < N_COARSE; ++i) {
    float sd = dco * sigma_co[ray * N_COARSE + i];
    cum += sd;
    float w = expf(-cum) * (1.0f - expf(-sd));
    int q = ray * N_COARSE + i;
    c0 += w * color_co[q * 3 + 0];
    c1 += w * color_co[q * 3 + 1];
    c2 += w * color_co[q * 3 + 2];
  }
  out[ray * 3 + 0] = c0; out[ray * 3 + 1] = c1; out[ray * 3 + 2] = c2;

  // fine pass: two-pointer merge of sorted coarse/fine sample lists
  float step = (ff - nf) / (float)(N_COARSE - 1);
  int i = 0, j = 0;
  auto pop = [&](float& t, float& sg, float* col) {
    float tc = (i < N_COARSE) ? nf + step * (float)i : 3.4e38f;
    float tf = (j < N_FINE) ? t_fi[ray * N_FINE + j] : 3.4e38f;
    if (tc <= tf) {
      int q = ray * N_COARSE + i; t = tc; sg = sigma_co[q];
      col[0] = color_co[q * 3 + 0]; col[1] = color_co[q * 3 + 1]; col[2] = color_co[q * 3 + 2];
      ++i;
    } else {
      int q = ray * N_FINE + j; t = tf; sg = sigma_fi[q];
      col[0] = color_fi[q * 3 + 0]; col[1] = color_fi[q * 3 + 1]; col[2] = color_fi[q * 3 + 2];
      ++j;
    }
  };
  float ct, cs, cc[3];
  pop(ct, cs, cc);
  cum = 0.0f;
  float f0 = 0.0f, f1 = 0.0f, f2 = 0.0f;
  const int NTOT = N_COARSE + N_FINE;
  for (int k = 0; k < NTOT; ++k) {
    float nt = 0.0f, ns = 0.0f, nc[3] = {0.0f, 0.0f, 0.0f};
    float delta;
    if (k < NTOT - 1) { pop(nt, ns, nc); delta = nt - ct; }
    else              { delta = 1e-4f; }
    float sd = delta * cs;
    cum += sd;
    float w = expf(-cum) * (1.0f - expf(-sd));
    f0 += w * cc[0]; f1 += w * cc[1]; f2 += w * cc[2];
    ct = nt; cs = ns; cc[0] = nc[0]; cc[1] = nc[1]; cc[2] = nc[2];
  }
  out[3 * N_RAYS + ray * 3 + 0] = f0;
  out[3 * N_RAYS + ray * 3 + 1] = f1;
  out[3 * N_RAYS + ray * 3 + 2] = f2;
}

// ---------------- launch ----------------------------------------------------
extern "C" void kernel_launch(void* const* d_in, const int* in_sizes, int n_in,
                              void* d_out, int out_size, void* d_ws, size_t ws_size,
                              hipStream_t stream) {
  (void)in_sizes; (void)n_in; (void)out_size; (void)ws_size;
  const float* hor   = (const float*)d_in[0];
  const float* ver   = (const float*)d_in[1];
  const float* trans = (const float*)d_in[2];
  const float* W_in  = (const float*)d_in[3];
  const float* b_in  = (const float*)d_in[4];
  const float* W_hid = (const float*)d_in[5];
  const float* b_hid = (const float*)d_in[6];
  const float* W_out = (const float*)d_in[7];
  const float* b_out = (const float*)d_in[8];
  const float* W_sig = (const float*)d_in[9];
  const float* b_sig = (const float*)d_in[10];
  const float* W_col = (const float*)d_in[11];
  const float* b_col = (const float*)d_in[12];
  const int*   nearp = (const int*)d_in[13];
  const int*   farp  = (const int*)d_in[14];

  const int NPTS_CO = N_RAYS * N_COARSE;   // 65536
  const int NPTS_FI = N_RAYS * N_FINE;     // 131072

  char* ws = (char*)d_ws;
  auto suballoc = [&](size_t bytes) -> char* {
    char* p = ws;
    ws += (bytes + 255) & ~(size_t)255;
    return p;
  };
  _Float16* wh      = (_Float16*)suballoc((size_t)NW_TOTAL * 2);
  _Float16* enc     = (_Float16*)suballoc((size_t)NPTS_FI * 64 * 2);
  float* sigma_co   = (float*)suballoc((size_t)NPTS_CO * 4);
  float* color_co   = (float*)suballoc((size_t)NPTS_CO * 12);
  float* sigma_fi   = (float*)suballoc((size_t)NPTS_FI * 4);
  float* color_fi   = (float*)suballoc((size_t)NPTS_FI * 12);
  float* t_fi       = (float*)suballoc((size_t)N_RAYS * N_FINE * 4);

  const size_t lds_bytes =
      ((size_t)LAYER_W * LDSK + 2 * (size_t)MTILE * LDSK) * sizeof(_Float16); // ~264KB

  convert_w_kernel<<<(NW_TOTAL + 255) / 256, 256, 0, stream>>>(W_in, W_hid, W_out, wh);

  prep_enc_kernel<<<(NPTS_CO + 255) / 256, 256, 0, stream>>>(
      hor, ver, trans, nullptr, nearp, farp, enc, N_COARSE, NPTS_CO);
  mlp_kernel<<<NPTS_CO / MTILE, 256, lds_bytes, stream>>>(
      enc, wh, b_in, b_hid, b_out, W_sig, b_sig, W_col, b_col, trans,
      sigma_co, color_co, NPTS_CO);

  resample_kernel<<<N_RAYS / 256, 256, 0, stream>>>(sigma_co, nearp, farp, t_fi);

  prep_enc_kernel<<<(NPTS_FI + 255) / 256, 256, 0, stream>>>(
      hor, ver, trans, t_fi, nearp, farp, enc, N_FINE, NPTS_FI);
  mlp_kernel<<<NPTS_FI / MTILE, 256, lds_bytes, stream>>>(
      enc, wh, b_in, b_hid, b_out, W_sig, b_sig, W_col, b_col, trans,
      sigma_fi, color_fi, NPTS_FI);

  render_kernel<<<N_RAYS / 256, 256, 0, stream>>>(
      sigma_co, color_co, sigma_fi, color_fi, t_fi, nearp, farp, (float*)d_out);
}
